// Ch_Pos_GCL_17772574671125
// MI455X (gfx1250) — matrix operand order repaired and checked by reading, not compiled, not run on hardware
//
#include <hip/hip_runtime.h>
#include <hip/hip_bf16.h>

typedef __attribute__((ext_vector_type(16))) __bf16 v16bf;
typedef __attribute__((ext_vector_type(8)))  float  v8f;
typedef unsigned short u16;

#define HID   128
#define KPAD  544   // 528 padded to multiple of 32
#define ASTR  552   // LDS row stride (bf16 elems) for the 528-wide A panel
#define CSTR  136   // LDS row stride (bf16 elems) for 128-wide activations
#define EPB   64    // edges per block
#define NWAVE 8     // 256 threads

// ---------- helpers ----------
__device__ __forceinline__ u16 f2bf(float f) {
    unsigned u = __float_as_uint(f);
    u += 0x7FFFu + ((u >> 16) & 1u);          // round-to-nearest-even
    return (u16)(u >> 16);
}
__device__ __forceinline__ unsigned pack2(float a, float b) {
    return (unsigned)f2bf(a) | ((unsigned)f2bf(b) << 16);
}
__device__ __forceinline__ float silu_f(float x) { return x / (1.f + __expf(-x)); }

// B fragment: lane holds column n; v0..v3 = K k0..k0+7, v4..v7 = K k0+16..k0+23
__device__ __forceinline__ v16bf ld_bfrag(const u16* p) {
    union { uint4 q[2]; v16bf v; } u;
    u.q[0] = *(const uint4*)p;
    u.q[1] = *(const uint4*)(p + 16);
    return u.v;
}
// A fragment: lane holds row m; v0..v3 = K k0..k0+7, v4..v7 = K k0+16..k0+23
__device__ __forceinline__ v16bf ld_afrag(const u16* p) {
    union { uint4 q[2]; v16bf v; } u;
    u.q[0] = *(const uint4*)p;
    u.q[1] = *(const uint4*)(p + 16);
    return u.v;
}

// 4 M-tiles x 1 N-tile: B loaded once per k-step, used by 4 WMMAs.
__device__ __forceinline__ void gemm_quad(const u16* A, int mstride, const u16* B,
                                          int ksteps, v8f (&c)[4]) {
    for (int kk = 0; kk < ksteps; ++kk) {
        v16bf b = ld_bfrag(B + kk * 32);
        #pragma unroll
        for (int mt = 0; mt < 4; ++mt) {
            v16bf a = ld_afrag(A + mt * mstride + kk * 32);
            c[mt] = __builtin_amdgcn_wmma_f32_16x16x32_bf16(false, a, false, b,
                                                            (short)0, c[mt], false, false);
        }
    }
}

// ---------- weight prep: f32 row-major [K,N] -> bf16 transposed [N][Kpad] ----------
__global__ void prep_weights(const float* __restrict__ ce_w1, const float* __restrict__ ce_w2,
                             const float* __restrict__ pe_w2, const float* __restrict__ sh_w,
                             u16* __restrict__ ws) {
    const int T1 = HID * KPAD;        // 69632
    const int TM = HID * HID;         // 16384
    const int total = T1 + 3 * TM;
    for (int idx = blockIdx.x * blockDim.x + threadIdx.x; idx < total;
         idx += gridDim.x * blockDim.x) {
        if (idx < T1) {
            int n = idx / KPAD, k = idx % KPAD;
            ws[idx] = (k < 528) ? f2bf(ce_w1[k * HID + n]) : (u16)0;
        } else {
            int r = idx - T1, mat = r / TM, t = r % TM;
            int n = t / HID, k = t % HID;
            const float* w = (mat == 0) ? ce_w2 : (mat == 1) ? pe_w2 : sh_w;
            ws[idx] = f2bf(w[k * HID + n]);
        }
    }
}

// ---------- fused edge kernel: 1 block = 64 edges; 8 waves x (1 N-tile, 4 M-tiles) ----
__global__ __launch_bounds__(256) void egcl_fused(
    const float* __restrict__ h, const float* __restrict__ coord, const int* __restrict__ edg,
    const float* __restrict__ nvecs, const float* __restrict__ na, const float* __restrict__ ea,
    const float* __restrict__ ce_b1, const float* __restrict__ ce_b2,
    const float* __restrict__ pe_w1, const float* __restrict__ pe_b1, const float* __restrict__ pe_b2,
    const float* __restrict__ sh_b, const float* __restrict__ att_w,
    const u16* __restrict__ WT1, const u16* __restrict__ WT2,
    const u16* __restrict__ WT3, const u16* __restrict__ WT4,
    float* __restrict__ out_p, float* __restrict__ chem_p, float* __restrict__ pos_p,
    float* __restrict__ cd_p, int E)
{
    // Abuf (70.6 KB) holds the gathered bf16 A panel during phase 0/1, then is
    // re-used as a 64x128 f32 buffer for the fused output across the att barrier.
    __shared__ __align__(16) u16  Abuf[EPB * ASTR];
    __shared__ u16  act1[EPB * CSTR];    // chem hidden       (17.4 KB)
    __shared__ u16  act2[EPB * CSTR];    // chem out (bf16)   (17.4 KB)
    __shared__ u16  posh[EPB * CSTR];    // pos hidden        (17.4 KB)
    __shared__ float nprod_sh[EPB], radial_sh[EPB], attsum_sh[EPB];

    const int tid = threadIdx.x;
    const int e0  = blockIdx.x * EPB;

    // ---- Phase 0: gather 64 edges into the A panel; geometry scalars ----
    {
        int er = tid >> 2, sub = tid & 3;      // 4 threads per edge
        int e = e0 + er;
        int row = edg[e], col = edg[E + e];
        const float4* hr = (const float4*)(h  + (size_t)row * HID);
        const float4* hc = (const float4*)(h  + (size_t)col * HID);
        const float4* nr = (const float4*)(na + (size_t)row * HID);
        const float4* nc = (const float4*)(na + (size_t)col * HID);
        u16* arow = &Abuf[er * ASTR];
        #pragma unroll
        for (int j = 0; j < 8; ++j) {
            int c = sub * 32 + j * 4;
            float4 v;
            v = hr[sub * 8 + j]; *(uint2*)&arow[      c] = make_uint2(pack2(v.x, v.y), pack2(v.z, v.w));
            v = hc[sub * 8 + j]; *(uint2*)&arow[128 + c] = make_uint2(pack2(v.x, v.y), pack2(v.z, v.w));
            v = nr[sub * 8 + j]; *(uint2*)&arow[256 + c] = make_uint2(pack2(v.x, v.y), pack2(v.z, v.w));
            v = nc[sub * 8 + j]; *(uint2*)&arow[384 + c] = make_uint2(pack2(v.x, v.y), pack2(v.z, v.w));
        }
        {
            float4 v = ((const float4*)(ea + (size_t)e * 16))[sub];
            *(uint2*)&arow[512 + sub * 4] = make_uint2(pack2(v.x, v.y), pack2(v.z, v.w));
        }
        *(uint2*)&arow[528 + sub * 4] = make_uint2(0u, 0u);   // K padding
        if (sub == 0) {
            float cx = coord[(size_t)row * 3 + 0] - coord[(size_t)col * 3 + 0];
            float cy = coord[(size_t)row * 3 + 1] - coord[(size_t)col * 3 + 1];
            float cz = coord[(size_t)row * 3 + 2] - coord[(size_t)col * 3 + 2];
            float rad = cx * cx + cy * cy + cz * cz;
            float inv = 1.f / (sqrtf(rad) + 1e-8f);
            cd_p[(size_t)e * 3 + 0] = cx * inv;
            cd_p[(size_t)e * 3 + 1] = cy * inv;
            cd_p[(size_t)e * 3 + 2] = cz * inv;
            float np = nvecs[(size_t)row * 3 + 0] * nvecs[(size_t)col * 3 + 0]
                     + nvecs[(size_t)row * 3 + 1] * nvecs[(size_t)col * 3 + 1]
                     + nvecs[(size_t)row * 3 + 2] * nvecs[(size_t)col * 3 + 2];
            nprod_sh[er] = np; radial_sh[er] = rad; attsum_sh[er] = 0.f;
        }
    }
    __syncthreads();

    // ---- Phase 1a: pos hidden layer (2->128, VALU) ----
    {
        int n = tid & 127, half = tid >> 7;
        float w0 = pe_w1[n], w1 = pe_w1[HID + n], b = pe_b1[n];
        #pragma unroll
        for (int r = 0; r < 32; ++r) {
            int row = half * 32 + r;
            float x = nprod_sh[row] * w0 + radial_sh[row] * w1 + b;
            posh[row * CSTR + n] = f2bf(silu_f(x));
        }
    }

    const int lane = tid & 31, wv = tid >> 5;
    const int hi = lane >> 4, lr = lane & 15;
    const int n = wv * 16 + lr;          // this wave owns columns wv*16..wv*16+15
    const int mb = hi * 8;

    // ---- Phase 1b: chem layer1 (528->128) ----
    {
        v8f c[4] = {};
        gemm_quad(&Abuf[lr * ASTR + hi * 8], 16 * ASTR,
                  WT1 + (size_t)n * KPAD + hi * 8, 17, c);
        float bb = ce_b1[n];
        #pragma unroll
        for (int mt = 0; mt < 4; ++mt)
            #pragma unroll
            for (int i = 0; i < 8; ++i)
                act1[(mt * 16 + mb + i) * CSTR + n] = f2bf(silu_f(c[mt][i] + bb));
    }
    __syncthreads();   // act1 ready; Abuf (A panel) now dead

    // ---- Phase 2: chem layer2 + pos layer2 ----
    {
        v8f c[4] = {};
        gemm_quad(&act1[lr * CSTR + hi * 8], 16 * CSTR, WT2 + n * HID + hi * 8, 4, c);
        float bb = ce_b2[n];
        #pragma unroll
        for (int mt = 0; mt < 4; ++mt)
            #pragma unroll
            for (int i = 0; i < 8; ++i) {
                float s = silu_f(c[mt][i] + bb);
                chem_p[(size_t)(e0 + mt * 16 + mb + i) * HID + n] = s;
                act2[(mt * 16 + mb + i) * CSTR + n] = f2bf(s);
            }
    }
    v8f p[4] = {};
    {
        v8f c[4] = {};
        gemm_quad(&posh[lr * CSTR + hi * 8], 16 * CSTR, WT3 + n * HID + hi * 8, 4, c);
        float bb = pe_b2[n];
        #pragma unroll
        for (int mt = 0; mt < 4; ++mt)
            #pragma unroll
            for (int i = 0; i < 8; ++i) {
                p[mt][i] = silu_f(c[mt][i] + bb);
                pos_p[(size_t)(e0 + mt * 16 + mb + i) * HID + n] = p[mt][i];
            }
    }
    __syncthreads();   // act2 ready

    // ---- Phase 3: share layer + fuse; spill fused rows to LDS; att partials ----
    float* ofs = reinterpret_cast<float*>(Abuf);   // 64 x 128 f32 (32 KB), reuses A panel
    {
        v8f c[4] = {};
        gemm_quad(&act2[lr * CSTR + hi * 8], 16 * CSTR, WT4 + n * HID + hi * 8, 4, c);
        float bb = sh_b[n];
        float aw = att_w[n];
        #pragma unroll
        for (int mt = 0; mt < 4; ++mt) {
            #pragma unroll
            for (int i = 0; i < 8; ++i) {
                float o = silu_f(c[mt][i] + bb) * p[mt][i];
                ofs[(mt * 16 + mb + i) * HID + n] = o;     // park across the barrier
                float part = o * aw;
                part += __shfl_xor(part, 1, 32);   // reduce over the 16-lane half (N dim)
                part += __shfl_xor(part, 2, 32);
                part += __shfl_xor(part, 4, 32);
                part += __shfl_xor(part, 8, 32);
                if (lr == 0) atomicAdd(&attsum_sh[mt * 16 + mb + i], part);
            }
        }
    }
    __syncthreads();

    // ---- Phase 4: sigmoid gate + final store ----
    #pragma unroll
    for (int mt = 0; mt < 4; ++mt)
        #pragma unroll
        for (int i = 0; i < 8; ++i) {
            int m = mt * 16 + mb + i;
            float g = 1.f / (1.f + __expf(-attsum_sh[m]));
            out_p[(size_t)(e0 + m) * HID + n] = ofs[m * HID + n] * g;
        }
}

extern "C" void kernel_launch(void* const* d_in, const int* in_sizes, int n_in,
                              void* d_out, int out_size, void* d_ws, size_t ws_size,
                              hipStream_t stream) {
    const float* h      = (const float*)d_in[0];
    const float* coord  = (const float*)d_in[1];
    const int*   edg    = (const int*)d_in[2];
    const float* nvecs  = (const float*)d_in[3];
    const float* na     = (const float*)d_in[4];
    const float* ea     = (const float*)d_in[5];
    const float* ce_w1  = (const float*)d_in[6];
    const float* ce_b1  = (const float*)d_in[7];
    const float* ce_w2  = (const float*)d_in[8];
    const float* ce_b2  = (const float*)d_in[9];
    const float* pe_w1  = (const float*)d_in[10];
    const float* pe_b1  = (const float*)d_in[11];
    const float* pe_w2  = (const float*)d_in[12];
    const float* pe_b2  = (const float*)d_in[13];
    const float* sh_w   = (const float*)d_in[14];
    const float* sh_b   = (const float*)d_in[15];
    const float* att_w  = (const float*)d_in[16];

    const int E = in_sizes[2] / 2;

    u16* ws = (u16*)d_ws;
    const u16* WT1 = ws;                         // 128 x 544 bf16
    const u16* WT2 = ws + HID * KPAD;            // 128 x 128
    const u16* WT3 = WT2 + HID * HID;            // 128 x 128
    const u16* WT4 = WT3 + HID * HID;            // 128 x 128

    float* out_p  = (float*)d_out;
    float* chem_p = out_p  + (size_t)E * HID;
    float* pos_p  = chem_p + (size_t)E * HID;
    float* cd_p   = pos_p  + (size_t)E * HID;

    prep_weights<<<232, 512, 0, stream>>>(ce_w1, ce_w2, pe_w2, sh_w, ws);
    egcl_fused<<<E / EPB, 256, 0, stream>>>(h, coord, edg, nvecs, na, ea,
                                            ce_b1, ce_b2, pe_w1, pe_b1, pe_b2,
                                            sh_b, att_w, WT1, WT2, WT3, WT4,
                                            out_p, chem_p, pos_p, cd_p, E);
}
